// Attention_919123001813
// MI455X (gfx1250) — matrix operand order repaired and checked by reading, not compiled, not run on hardware
//
#include <hip/hip_runtime.h>

// ---------------------------------------------------------------------------
// Attention (B=4, N=2048, DIM=1024, H=16, HD=64) for gfx1250 (CDNA5, wave32)
// BF16 WMMA pipeline: cvt -> QKV GEMM -> flash attention -> proj GEMM + bias
// Fragment-ready LDS layouts (all fragment loads = 2x contiguous ds_load_b128),
// async global->LDS staging (global_load_async_to_lds_b128) with 2-stage
// LDS double buffering: tile i+1 streams in while tile i runs on the WMMA pipe.
// ---------------------------------------------------------------------------

typedef __attribute__((ext_vector_type(16))) __bf16 v16bf;
typedef __attribute__((ext_vector_type(8)))  float  v8f;
typedef __attribute__((ext_vector_type(4)))  unsigned int uint4v;
typedef __attribute__((__vector_size__(16))) int v4i_t;
typedef unsigned long long u64;
typedef unsigned short     u16;

#define BB   4
#define SEQ  2048
#define DIM  1024
#define NH   16
#define HD   64
#define SCALE 0.125f   // 1/sqrt(64)

union BfFrag { v16bf v; u16 u[16]; };

__device__ inline u16 f2bf(float f) {
  unsigned u = __builtin_bit_cast(unsigned, f);
  u += 0x7FFFu + ((u >> 16) & 1u);      // round-to-nearest-even
  return (u16)(u >> 16);
}

__device__ inline v8f wmma_bf16(const BfFrag& a, const BfFrag& b, v8f c) {
  return __builtin_amdgcn_wmma_f32_16x16x32_bf16(
      false, a.v, false, b.v, (short)0, c, false, false);
}

// ---- CDNA5 async global->LDS copy (16B), guarded fallback to plain copy ----
__device__ inline void async_copy16(const u16* g, u16* l) {
#if __has_builtin(__builtin_amdgcn_global_load_async_to_lds_b128)
  __builtin_amdgcn_global_load_async_to_lds_b128(
      (__attribute__((address_space(1))) v4i_t*)(u16*)g,
      (__attribute__((address_space(3))) v4i_t*)l, 0, 0);
#else
  *(uint4v*)l = *(const uint4v*)g;
#endif
}

__device__ inline void async_join() {
#if __has_builtin(__builtin_amdgcn_global_load_async_to_lds_b128)
#if __has_builtin(__builtin_amdgcn_s_wait_asynccnt)
  __builtin_amdgcn_s_wait_asynccnt(0);
#else
  asm volatile("s_wait_asynccnt 0" ::: "memory");
#endif
#endif
}

__device__ inline void dscnt_wait0() {
#if __has_builtin(__builtin_amdgcn_s_wait_dscnt)
  __builtin_amdgcn_s_wait_dscnt(0);
#else
  asm volatile("s_wait_dscnt 0" ::: "memory");
#endif
}

// ---------------------------------------------------------------------------
// fp32 -> bf16 conversion
// ---------------------------------------------------------------------------
__global__ __launch_bounds__(256) void attn_cvt_bf16(
    const float* __restrict__ src, u16* __restrict__ dst, int n) {
  int i = blockIdx.x * 256 + threadIdx.x;
  if (i < n) dst[i] = f2bf(src[i]);
}

// ---------------------------------------------------------------------------
// GEMM staging helpers (256-thread blocks)
// ---------------------------------------------------------------------------
__device__ inline void gemm_stage_a(const u16* __restrict__ A, int mBase, int K,
                                    int k0, u16* lAbuf, int t) {
  #pragma unroll
  for (int i = 0; i < 4; ++i) {
    int c   = t + 256 * i;            // 1024 16B chunks (128x64 tile)
    int row = c >> 3;
    int col = (c & 7) << 3;
    async_copy16(&A[(size_t)(mBase + row) * K + k0 + col], &lAbuf[row * 64 + col]);
  }
}

__device__ inline void gemm_load_b(const u16* __restrict__ Bm, int N, int k0,
                                   int nBase, int t, u64 g[2][4]) {
  #pragma unroll
  for (int i = 0; i < 2; ++i) {
    int blk = t + 256 * i;            // 512 4x4 blocks (16 k-blk x 32 n-blk)
    int bk  = (blk & 15) << 2;
    int bn  = (blk >> 4) << 2;
    #pragma unroll
    for (int j = 0; j < 4; ++j)
      g[i][j] = *(const u64*)&Bm[(size_t)(k0 + bk + j) * N + nBase + bn];
  }
}

__device__ inline void gemm_store_b(const u64 g[2][4], u16* lBbuf, int t) {
  #pragma unroll
  for (int i = 0; i < 2; ++i) {
    int blk = t + 256 * i;
    int bk  = (blk & 15) << 2;
    int bn  = (blk >> 4) << 2;
    #pragma unroll
    for (int jn = 0; jn < 4; ++jn) {  // 4x4 u16 micro-transpose -> lB[n][k]
      u64 o = 0;
      #pragma unroll
      for (int jk = 0; jk < 4; ++jk)
        o |= (u64)((u16)(g[i][jk] >> (16 * jn))) << (16 * jk);
      *(u64*)&lBbuf[(size_t)(bn + jn) * 64 + bk] = o;
    }
  }
}

// ---------------------------------------------------------------------------
// BF16 GEMM: C[M,N] = A[M,K] @ B[K,N], f32 accumulate.
// Block tile 128x128, K-chunk 64, double-buffered LDS; 8 waves in 4(M) x 2(N);
// wave tile 32x64 = 2x4 WMMA accumulators. B staged transposed: lBt[n][k].
// mode 0: scatter into per-head Q/K/V bf16 [B,H,SEQ,HD]; mode 1: f32 + bias.
// ---------------------------------------------------------------------------
__global__ __launch_bounds__(256) void attn_gemm_bf16(
    const u16* __restrict__ A, const u16* __restrict__ Bm,
    int M, int N, int K, int mode,
    u16* __restrict__ q_out, u16* __restrict__ k_out, u16* __restrict__ v_out,
    float* __restrict__ f_out, const float* __restrict__ bias) {
  __shared__ alignas(16) u16 lA[2][128 * 64];    // [m][k]
  __shared__ alignas(16) u16 lBt[2][128 * 64];   // [n][k]

  const int t    = threadIdx.x;
  const int lane = t & 31;
  const int wave = t >> 5;
  const int wm   = wave >> 1;        // 0..3 : 32 rows each
  const int wn   = wave & 1;         // 0..1 : 64 cols each
  const int hl   = lane >> 4;
  const int l16  = lane & 15;

  const int mBase = blockIdx.y * 128;
  const int nBase = blockIdx.x * 128;
  const int niter = K >> 6;

  v8f acc[2][4] = {};
  u64 gB[2][4];

  // ---- prologue: stage tile 0 ----
  gemm_stage_a(A, mBase, K, 0, lA[0], t);
  gemm_load_b(Bm, N, 0, nBase, t, gB);
  gemm_store_b(gB, lBt[0], t);
  async_join();
  __syncthreads();

  for (int it = 0; it < niter; ++it) {
    const int cur  = it & 1;
    const bool more = (it + 1) < niter;
    if (more) {                        // issue next tile's traffic first
      gemm_stage_a(A, mBase, K, (it + 1) << 6, lA[1 - cur], t);
      gemm_load_b(Bm, N, (it + 1) << 6, nBase, t, gB);
    }

    // ---- compute current tile: 2 k-subchunks of 32, 2x4 wmma each ----
    const u16* lAc = lA[cur];
    const u16* lBc = lBt[cur];
    #pragma unroll
    for (int s = 0; s < 2; ++s) {
      BfFrag af[2], bfr[4];
      #pragma unroll
      for (int mt = 0; mt < 2; ++mt) {
        const u16* p = &lAc[(size_t)(wm * 32 + mt * 16 + l16) * 64 + s * 32 + 16 * hl];
        *(uint4v*)&af[mt].u[0] = *(const uint4v*)p;         // k = 16h..16h+7
        *(uint4v*)&af[mt].u[8] = *(const uint4v*)(p + 8);   // k = 16h+8..
      }
      #pragma unroll
      for (int nt = 0; nt < 4; ++nt) {
        const u16* p = &lBc[(size_t)(wn * 64 + nt * 16 + l16) * 64 + s * 32 + 16 * hl];
        *(uint4v*)&bfr[nt].u[0] = *(const uint4v*)p;        // k = 16h..16h+7
        *(uint4v*)&bfr[nt].u[8] = *(const uint4v*)(p + 8);  // k = 16h+8..
      }
      #pragma unroll
      for (int mt = 0; mt < 2; ++mt)
        #pragma unroll
        for (int nt = 0; nt < 4; ++nt)
          acc[mt][nt] = wmma_bf16(af[mt], bfr[nt], acc[mt][nt]);
    }

    if (more) {
      gemm_store_b(gB, lBt[1 - cur], t);   // transpose after compute
      async_join();
      __syncthreads();                     // next tile ready; cur reads done
    }
  }

  // ---- epilogue: C element r is (m = r + 8*hl, n = l16) in each tile ----
  #pragma unroll
  for (int mt = 0; mt < 2; ++mt) {
    #pragma unroll
    for (int nt = 0; nt < 4; ++nt) {
      #pragma unroll
      for (int r = 0; r < 8; ++r) {
        int m = mBase + wm * 32 + mt * 16 + r + 8 * hl;
        int n = nBase + wn * 64 + nt * 16 + l16;
        float val = acc[mt][nt][r];
        if (mode == 0) {
          int which  = n >> 10;          // 0=q 1=k 2=v
          int within = n & 1023;
          int h = within >> 6, d = within & 63;
          int b = m >> 11, nn = m & 2047;
          u16* dst = (which == 0) ? q_out : (which == 1) ? k_out : v_out;
          dst[(((size_t)b * NH + h) * SEQ + nn) * HD + d] = f2bf(val);
        } else {
          f_out[(size_t)m * N + n] = val + bias[n];
        }
      }
    }
  }
}

// ---------------------------------------------------------------------------
// Flash staging helpers (128-thread blocks)
// ---------------------------------------------------------------------------
__device__ inline void flash_stage_k(const u16* __restrict__ Kq, size_t kvBase,
                                     u16* lKbuf, int t) {
  #pragma unroll
  for (int i = 0; i < 4; ++i) {
    int c   = t + 128 * i;            // 512 16B chunks (128 rows x 4, 16-col stride)
    int row = c >> 2;
    int col = (c & 3) << 4;
    async_copy16(&Kq[kvBase + (size_t)row * 64 + col], &lKbuf[row * 64 + col]);
  }
}

__device__ inline void flash_load_v(const u16* __restrict__ V, size_t kvBase,
                                    int t, u64 g[2][4]) {
  #pragma unroll
  for (int i = 0; i < 2; ++i) {
    int blk  = t + 128 * i;           // 256 4x4 blocks (16 key-blk x 16 d-blk)
    int bkey = (blk & 15) << 2;
    int bd   = (blk >> 4) << 2;
    #pragma unroll
    for (int j = 0; j < 4; ++j)
      g[i][j] = *(const u64*)&V[kvBase + (size_t)(bkey + j) * 64 + bd];
  }
}

__device__ inline void flash_store_v(const u64 g[2][4], u16* lVbuf, int t) {
  #pragma unroll
  for (int i = 0; i < 2; ++i) {
    int blk  = t + 128 * i;
    int bkey = (blk & 15) << 2;
    int bd   = (blk >> 4) << 2;
    #pragma unroll
    for (int jd = 0; jd < 4; ++jd) {  // micro-transpose -> lVt[d][key]
      u64 o = 0;
      #pragma unroll
      for (int jk = 0; jk < 4; ++jk)
        o |= (u64)((u16)(g[i][jk] >> (16 * jd))) << (16 * jk);
      *(u64*)&lVbuf[(size_t)(bd + jd) * 64 + bkey] = o;
    }
  }
}

// ---------------------------------------------------------------------------
// Flash attention: grid (32 qblocks, 64 bh), 128 threads = 4 waves.
// Each wave owns a 16-row Q strip; 64-key K/V tiles, double-buffered.
// ---------------------------------------------------------------------------
__global__ __launch_bounds__(128) void attn_flash_bf16(
    const u16* __restrict__ Q, const u16* __restrict__ Kq,
    const u16* __restrict__ V, u16* __restrict__ Oattn) {
  __shared__ alignas(16) u16 ldsK[2][64 * 64];     // [key][d]
  __shared__ alignas(16) u16 ldsVt[2][64 * 64];    // [d][key]
  __shared__ alignas(16) u16 ldsP[4 * 16 * 64];    // per-wave [m][key]

  const int t    = threadIdx.x;
  const int lane = t & 31;
  const int wave = t >> 5;
  const int hl   = lane >> 4;
  const int l16  = lane & 15;

  const int qblk  = blockIdx.x;      // 0..31
  const int bh    = blockIdx.y;      // 0..63
  const int qrow0 = qblk * 64 + wave * 16;
  const size_t bhBase = (size_t)bh * SEQ * HD;

  // ---- Q strip: two 16x32 A-fragments, vectorized 16B loads ----
  BfFrag aq[2];
  {
    const u16* qp = Q + bhBase + (size_t)(qrow0 + l16) * HD;
    #pragma unroll
    for (int f = 0; f < 2; ++f) {
      const u16* p = qp + f * 32 + 16 * hl;
      *(uint4v*)&aq[f].u[0] = *(const uint4v*)p;
      *(uint4v*)&aq[f].u[8] = *(const uint4v*)(p + 8);
    }
  }

  v8f oacc[4] = {};
  float rmax[8], rsum[8];
  #pragma unroll
  for (int r = 0; r < 8; ++r) { rmax[r] = -3.0e38f; rsum[r] = 0.0f; }

  u64 gV[2][4];
  // ---- prologue: stage tile 0 ----
  flash_stage_k(Kq, bhBase, ldsK[0], t);
  flash_load_v(V, bhBase, t, gV);
  flash_store_v(gV, ldsVt[0], t);
  async_join();
  __syncthreads();

  const int niter = SEQ / 64;
  for (int kt = 0; kt < niter; ++kt) {
    const int cur   = kt & 1;
    const bool more = (kt + 1) < niter;
    if (more) {                        // issue next tile's traffic first
      const size_t nb = bhBase + (size_t)(kt + 1) * 64 * HD;
      flash_stage_k(Kq, nb, ldsK[1 - cur], t);
      flash_load_v(V, nb, t, gV);
    }

    // ---- S = Q K^T : four 16x16 score tiles (64 keys) ----
    const u16* lKc = ldsK[cur];
    v8f s4[4] = {};
    #pragma unroll
    for (int f = 0; f < 2; ++f) {
      BfFrag bk[4];
      #pragma unroll
      for (int kb = 0; kb < 4; ++kb) {
        const u16* p = &lKc[(size_t)(kb * 16 + l16) * 64 + f * 32 + 16 * hl];
        *(uint4v*)&bk[kb].u[0] = *(const uint4v*)p;
        *(uint4v*)&bk[kb].u[8] = *(const uint4v*)(p + 8);
      }
      #pragma unroll
      for (int kb = 0; kb < 4; ++kb)
        s4[kb] = wmma_bf16(aq[f], bk[kb], s4[kb]);
    }

    // ---- online softmax; row m = r + 8*hl across 16 lanes of this half ----
    #pragma unroll
    for (int r = 0; r < 8; ++r) {
      float v0 = s4[0][r] * SCALE, v1 = s4[1][r] * SCALE;
      float v2 = s4[2][r] * SCALE, v3 = s4[3][r] * SCALE;
      float mx = fmaxf(fmaxf(v0, v1), fmaxf(v2, v3));
      #pragma unroll
      for (int off = 1; off < 16; off <<= 1)
        mx = fmaxf(mx, __shfl_xor(mx, off, 32));
      float mnew  = fmaxf(rmax[r], mx);
      float alpha = __expf(rmax[r] - mnew);
      rmax[r] = mnew;
      float p0 = __expf(v0 - mnew), p1 = __expf(v1 - mnew);
      float p2 = __expf(v2 - mnew), p3 = __expf(v3 - mnew);
      float ps = (p0 + p1) + (p2 + p3);
      #pragma unroll
      for (int off = 1; off < 16; off <<= 1)
        ps += __shfl_xor(ps, off, 32);
      rsum[r] = rsum[r] * alpha + ps;
      #pragma unroll
      for (int tt = 0; tt < 4; ++tt) oacc[tt][r] *= alpha;
      int m = r + 8 * hl;
      u16* pp = &ldsP[(size_t)wave * 1024 + (size_t)m * 64 + l16];
      pp[0]  = f2bf(p0);
      pp[16] = f2bf(p1);
      pp[32] = f2bf(p2);
      pp[48] = f2bf(p3);
    }
    dscnt_wait0();   // same-wave LDS RAW on ldsP (DS ops are in-order)

    // ---- O += P @ V : two 16x32 P fragments x four 16-wide V slabs ----
    const u16* lVc = ldsVt[cur];
    #pragma unroll
    for (int s = 0; s < 2; ++s) {
      BfFrag ap;
      {
        const u16* p = &ldsP[(size_t)wave * 1024 + (size_t)l16 * 64 + s * 32 + 16 * hl];
        *(uint4v*)&ap.u[0] = *(const uint4v*)p;
        *(uint4v*)&ap.u[8] = *(const uint4v*)(p + 8);
      }
      #pragma unroll
      for (int nt = 0; nt < 4; ++nt) {
        BfFrag bv;
        const u16* p = &lVc[(size_t)(nt * 16 + l16) * 64 + s * 32 + 16 * hl];
        *(uint4v*)&bv.u[0] = *(const uint4v*)p;
        *(uint4v*)&bv.u[8] = *(const uint4v*)(p + 8);
        oacc[nt] = wmma_bf16(ap, bv, oacc[nt]);
      }
    }

    if (more) {
      flash_store_v(gV, ldsVt[1 - cur], t);  // transpose after compute
      async_join();
      __syncthreads();                       // next tile ready; cur reads done
    }
  }

  // ---- normalize + write bf16 attn output [B, SEQ, DIM] ----
  const int b = bh >> 4, h = bh & 15;
  #pragma unroll
  for (int nt = 0; nt < 4; ++nt) {
    #pragma unroll
    for (int r = 0; r < 8; ++r) {
      int m    = r + 8 * hl;
      int rowN = qrow0 + m;
      int col  = h * 64 + nt * 16 + l16;
      float val = oacc[nt][r] / rsum[r];
      Oattn[((size_t)b * SEQ + rowN) * DIM + col] = f2bf(val);
    }
  }
}

// ---------------------------------------------------------------------------
// launch
// ---------------------------------------------------------------------------
extern "C" void kernel_launch(void* const* d_in, const int* in_sizes, int n_in,
                              void* d_out, int out_size, void* d_ws, size_t ws_size,
                              hipStream_t stream) {
  const float* x      = (const float*)d_in[0];
  const float* w_qkv  = (const float*)d_in[1];
  const float* w_proj = (const float*)d_in[2];
  const float* b_proj = (const float*)d_in[3];
  float*       out    = (float*)d_out;

  const size_t nX  = (size_t)BB * SEQ * DIM;     // 8388608
  const size_t nWq = (size_t)DIM * 3 * DIM;      // 3145728
  const size_t nWp = (size_t)DIM * DIM;          // 1048576
  const size_t nH  = (size_t)BB * NH * SEQ * HD; // 8388608

  char* ws = (char*)d_ws;
  size_t off = 0;
  u16* xb  = (u16*)(ws + off); off += nX  * 2;
  u16* wqb = (u16*)(ws + off); off += nWq * 2;
  u16* wpb = (u16*)(ws + off); off += nWp * 2;
  u16* qb  = (u16*)(ws + off); off += nH  * 2;
  u16* kb  = (u16*)(ws + off); off += nH  * 2;
  u16* vb  = (u16*)(ws + off); off += nH  * 2;
  u16* ab  = (u16*)(ws + off); off += nX  * 2;   // attn output bf16 [B,SEQ,DIM]

  attn_cvt_bf16<<<dim3((unsigned)((nX  + 255) / 256)), 256, 0, stream>>>(x,      xb,  (int)nX);
  attn_cvt_bf16<<<dim3((unsigned)((nWq + 255) / 256)), 256, 0, stream>>>(w_qkv,  wqb, (int)nWq);
  attn_cvt_bf16<<<dim3((unsigned)((nWp + 255) / 256)), 256, 0, stream>>>(w_proj, wpb, (int)nWp);

  // QKV:  [8192,1024] x [1024,3072] -> scatter q/k/v per-head
  attn_gemm_bf16<<<dim3(3072 / 128, 8192 / 128), 256, 0, stream>>>(
      xb, wqb, 8192, 3072, 1024, 0, qb, kb, vb, nullptr, nullptr);

  // flash attention
  attn_flash_bf16<<<dim3(SEQ / 64, BB * NH), 128, 0, stream>>>(qb, kb, vb, ab);

  // projection: [8192,1024] x [1024,1024] + bias -> fp32 out
  attn_gemm_bf16<<<dim3(1024 / 128, 8192 / 128), 256, 0, stream>>>(
      ab, wpb, 8192, 1024, 1024, 1, nullptr, nullptr, nullptr, out, b_proj);
}
